// QuantitizedRCNN_24120536334895
// MI455X (gfx1250) — compile-verified, matrix operand order checked by reading, not run on hardware
//
#include <hip/hip_runtime.h>
#include <hip/hip_bf16.h>

// ---------------- feature detection ----------------
#if defined(__has_builtin)
#  if __has_builtin(__builtin_amdgcn_tensor_load_to_lds)
#    define HAVE_TDM 1
#  endif
#endif
#ifndef HAVE_TDM
#  define HAVE_TDM 0
#endif
#if defined(__has_include)
#  if __has_include(<hip/amd_detail/amd_gfx1250_TDM.h>)
#    define TDM_SIX_ARG 1
#  else
#    define TDM_SIX_ARG 0
#  endif
#else
#  define TDM_SIX_ARG 0
#endif

// ---------------- types / helpers ----------------
typedef __attribute__((ext_vector_type(16))) __bf16 v16bf;
typedef __attribute__((ext_vector_type(8)))  float  v8f;
typedef __attribute__((ext_vector_type(4)))  unsigned int u32x4;
typedef __attribute__((ext_vector_type(4)))  int i32x4;
typedef __attribute__((ext_vector_type(8)))  int i32x8;

struct U128x2 { uint4 lo, hi; };

// 16x32 bf16 WMMA fragment half-row: 8 bf16 at p, 8 bf16 at p+16 (K and K+16 chunks)
__device__ __forceinline__ v16bf mkfrag(const unsigned short* p) {
    U128x2 t;
    t.lo = *(const uint4*)(p);
    t.hi = *(const uint4*)(p + 16);
    return __builtin_bit_cast(v16bf, t);
}

__device__ __forceinline__ float bf2f(unsigned short s) {
    union { unsigned u; float f; } v; v.u = ((unsigned)s) << 16; return v.f;
}
__device__ __forceinline__ unsigned short f2bf(float f) {
    union { float f; unsigned u; } v; v.f = f;
    unsigned u = v.u;
    return (unsigned short)((u + 0x7FFFu + ((u >> 16) & 1u)) >> 16);
}
// quantifier with v = 64 (max|.| >= 1 at this workload's scale)
__device__ __forceinline__ float quant64(float x) {
    return truncf(x * 0.015625f) * 0.015625f;
}

#define CH 256
#define WIN_MAX (3 * 18 * 256)   // >= 4*10*256

// ---------------- input pad + quantize: NCHW f32 -> padded NHWC bf16 ----------------
__global__ void pad_quant_in_k(const float* __restrict__ x, unsigned short* __restrict__ dst) {
    const int H = 64, W = 64, Hp = 66, Wp = 66;
    long long idx = (long long)blockIdx.x * blockDim.x + threadIdx.x;
    long long total = 32LL * Hp * Wp * CH;
    if (idx >= total) return;
    int c = (int)(idx & (CH - 1));
    long long q = idx >> 8;
    int wp = (int)(q % Wp); q /= Wp;
    int hp = (int)(q % Hp);
    int n  = (int)(q / Hp);
    unsigned short out = 0;
    if (c < 3 && hp >= 1 && hp <= H && wp >= 1 && wp <= W) {
        float v = x[(((long long)n * 3 + c) * H + (hp - 1)) * W + (wp - 1)];
        out = f2bf(quant64(v));
    }
    dst[idx] = out;
}

// ---------------- weight swizzles: OIHW f32 -> OHWI bf16 ----------------
__global__ void convw_to_bf_k(const float* __restrict__ w, unsigned short* __restrict__ dst) {
    int idx = blockIdx.x * blockDim.x + threadIdx.x;
    const int total = CH * CH * 9;
    if (idx >= total) return;
    int i  = idx & (CH - 1);
    int q  = idx >> 8;
    int kw = q % 3; q /= 3;
    int kh = q % 3;
    int o  = q / 3;
    float v = w[(((long long)o * CH + i) * 3 + kh) * 3 + kw];
    dst[idx] = f2bf(v);
}
__global__ void f32_to_bf_k(const float* __restrict__ s, unsigned short* __restrict__ d, int n) {
    int idx = blockIdx.x * blockDim.x + threadIdx.x;
    if (idx < n) d[idx] = f2bf(s[idx]);
}

// ---------------- conv3x3 implicit GEMM + fused residual epilogue ----------------
// Layouts: activations padded NHWC bf16 [32][H+2][W+2][256]; weights OHWI bf16 [256][3][3][256].
// Block: 256 threads = 8 waves = 2 pixel tiles x 4 oc-quads. Each wave: 16 px x 64 oc, 288 WMMAs.
// A window (3x18x256 or 4x10x256 bf16) staged to LDS per pixel tile via TDM tensor_load_to_lds.
__global__ __launch_bounds__(256) void conv_wmma_k(
    const unsigned short* __restrict__ src,
    const unsigned short* __restrict__ wgt,
    const unsigned short* __restrict__ r0,
    const unsigned short* __restrict__ r1,
    const unsigned short* __restrict__ r2,
    const unsigned short* __restrict__ r3,
    const float* __restrict__ alphaT,    // alpha + t*6
    float* __restrict__ preBN,
    int lgW)                             // H == W == 1<<lgW
{
    const int W  = 1 << lgW;
    const int Wp = W + 2, Hp = W + 2;
    const int lane = threadIdx.x & 31;
    const int wv   = threadIdx.x >> 5;          // 0..7
    const int tsel = wv >> 2;                   // pixel tile within block
    const int ocq  = (wv & 3) << 6;             // 0,64,128,192
    const int pTile = (blockIdx.x << 1) + tsel;
    const int l15  = lane & 15;
    const int kOff = (lane >> 4) << 3;          // 0 or 8

    const bool small = (W < 16);                // W==8: 16-px tile spans 2 rows
    const int  C = small ? Wp : 18;             // window width (pixels)
    const int  R = small ? 4  : 3;              // window rows

    const int p0  = pTile << 4;
    const int n0  = p0 >> (2 * lgW);
    const int rem = p0 & ((1 << (2 * lgW)) - 1);
    const int h0  = rem >> lgW;
    const int w0  = small ? 0 : (rem & (W - 1));
    const unsigned short* gwin = src + ((((long long)n0 * Hp + h0) * Wp + w0) << 8);

    __shared__ unsigned short smem[2][WIN_MAX];
    unsigned short* win = smem[tsel];

#if HAVE_TDM
    if ((wv & 3) == 0) {                        // one wave per pixel tile drives the TDM
        unsigned lds_off = (unsigned)(unsigned long long)(&win[0]);
        unsigned long long ga = (unsigned long long)gwin;
        unsigned td0 = (unsigned)(C << 8);      // tensor_dim0 / tile_dim0 (elements)
        unsigned ts0 = (unsigned)(Wp << 8);     // row stride (elements)
        u32x4 g0;
        g0[0] = 1u;                                                   // count=1
        g0[1] = lds_off;                                              // lds_addr (bytes)
        g0[2] = (unsigned)(ga & 0xFFFFFFFFu);                         // global_addr lo
        g0[3] = (unsigned)((ga >> 32) & 0x01FFFFFFu) | (2u << 30);    // global_addr hi | type=2
        i32x8 g1;
        g1[0] = 0x00010000;                                           // data_size=1 (2B)
        g1[1] = (int)((td0 & 0xFFFFu) << 16);                         // tensor_dim0 lo16
        g1[2] = (int)((td0 >> 16) | (((unsigned)R & 0xFFFFu) << 16)); // td0 hi16 | tensor_dim1 lo16
        g1[3] = (int)(td0 << 16);                                     // tile_dim0 @ [31:16]
        g1[4] = (int)(R & 0xFFFF);                                    // tile_dim1=R, tile_dim2=0
        g1[5] = (int)ts0;                                             // tensor_dim0_stride lo32
        g1[6] = 0;
        g1[7] = 0;
        i32x4 z4 = {0, 0, 0, 0};
#if TDM_SIX_ARG
        i32x8 z8 = {0, 0, 0, 0, 0, 0, 0, 0};
        __builtin_amdgcn_tensor_load_to_lds(g0, g1, z4, z4, z8, 0);
#else
        __builtin_amdgcn_tensor_load_to_lds(g0, g1, z4, z4, 0);
#endif
#if defined(__has_builtin)
#if __has_builtin(__builtin_amdgcn_s_wait_tensorcnt)
        __builtin_amdgcn_s_wait_tensorcnt(0);
#endif
#endif
    }
    __syncthreads();
#else
    {   // cooperative fill (fallback / host pass): 4 waves per tile copy the window
        const int halfTid = threadIdx.x & 127;
        const int n128row = C << 5;             // uint4 per window row
        for (int r = 0; r < R; ++r) {
            const uint4* srow = (const uint4*)(gwin + (((long long)r * Wp) << 8));
            uint4* drow = (uint4*)(win + ((r * C) << 8));
            for (int i = halfTid; i < n128row; i += 128) drow[i] = srow[i];
        }
    }
    __syncthreads();
#endif

    // lane's A row within tile
    const int mRow = small ? (l15 >> 3) : 0;
    const int mCol = small ? (l15 & 7)  : l15;
    // lane's B columns (4 oc tiles)
    const int oc0 = ocq + l15;
    const unsigned short* bbase = wgt + (((long long)oc0 * 9) << 8) + kOff;
    const int OCT = (16 * 9) << 8;              // +16 oc in OHWI elements

    v8f acc0 = {}, acc1 = {}, acc2 = {}, acc3 = {};
    #pragma unroll
    for (int kh = 0; kh < 3; ++kh) {
        #pragma unroll
        for (int kw = 0; kw < 3; ++kw) {
            const unsigned short* ap = win + ((((mRow + kh) * C) + (mCol + kw)) << 8) + kOff;
            const unsigned short* bp = bbase + ((kh * 3 + kw) << 8);
            __builtin_prefetch(bp + CH, 0, 1);  // global_prefetch_b8 on weight stream
            #pragma unroll
            for (int kc = 0; kc < CH; kc += 32) {
                v16bf a  = mkfrag(ap + kc);     // ds_load_b128 x2
                v16bf b0 = mkfrag(bp + kc);
                v16bf b1 = mkfrag(bp + kc + OCT);
                v16bf b2 = mkfrag(bp + kc + 2 * OCT);
                v16bf b3 = mkfrag(bp + kc + 3 * OCT);
                acc0 = __builtin_amdgcn_wmma_f32_16x16x32_bf16(false, a, false, b0, (short)0, acc0, false, false);
                acc1 = __builtin_amdgcn_wmma_f32_16x16x32_bf16(false, a, false, b1, (short)0, acc1, false, false);
                acc2 = __builtin_amdgcn_wmma_f32_16x16x32_bf16(false, a, false, b2, (short)0, acc2, false, false);
                acc3 = __builtin_amdgcn_wmma_f32_16x16x32_bf16(false, a, false, b3, (short)0, acc3, false, false);
            }
        }
    }

    // Epilogue: a = alpha[0]*relu(conv) + sum_i alpha[2+i]*hist[i]
    const float a0 = alphaT[0];
    const float alr[4] = { alphaT[2], alphaT[3], alphaT[4], alphaT[5] };
    const unsigned short* hp[4] = { r0, r1, r2, r3 };
    const int rowM = kOff;                      // C/D row base for this lane half
    #pragma unroll
    for (int j = 0; j < 8; ++j) {
        int p  = p0 + rowM + j;
        int wq = p & (W - 1);
        int hq = (p >> lgW) & (W - 1);
        int nq = p >> (2 * lgW);
        long long padOff = ((((long long)nq * Hp + hq + 1) * Wp) + wq + 1) << 8;
        float v0 = a0 * fmaxf(acc0[j], 0.0f);
        float v1 = a0 * fmaxf(acc1[j], 0.0f);
        float v2 = a0 * fmaxf(acc2[j], 0.0f);
        float v3 = a0 * fmaxf(acc3[j], 0.0f);
        #pragma unroll
        for (int k = 0; k < 4; ++k) {
            if (hp[k]) {
                v0 += alr[k] * bf2f(hp[k][padOff + oc0]);
                v1 += alr[k] * bf2f(hp[k][padOff + oc0 + 16]);
                v2 += alr[k] * bf2f(hp[k][padOff + oc0 + 32]);
                v3 += alr[k] * bf2f(hp[k][padOff + oc0 + 48]);
            }
        }
        long long o = ((long long)p << 8) + oc0;
        preBN[o]      = v0;
        preBN[o + 16] = v1;
        preBN[o + 32] = v2;
        preBN[o + 48] = v3;
    }
}

// ---------------- BN stats: coalesced two-stage deterministic reduction ----------------
__global__ void bn_partial_k(const float* __restrict__ preBN, float* __restrict__ partial, int P) {
    const int c = threadIdx.x;                  // channel
    const int b = blockIdx.x;                   // 0..63
    const int stripe = (P + 63) >> 6;
    int pb = b * stripe;
    int pe = pb + stripe; if (pe > P) pe = P;
    float s = 0.f, sq = 0.f;
    for (int p = pb; p < pe; ++p) {
        float v = preBN[((long long)p << 8) + c];
        s += v; sq += v * v;
    }
    partial[b * 512 + c]       = s;
    partial[b * 512 + 256 + c] = sq;
}
__global__ void bn_finalize_k(const float* __restrict__ partial, float* __restrict__ stats, int P) {
    const int c = threadIdx.x;                  // 256 threads
    float s = 0.f, sq = 0.f;
    for (int b = 0; b < 64; ++b) {
        s  += partial[b * 512 + c];
        sq += partial[b * 512 + 256 + c];
    }
    float mean = s / (float)P;
    float var  = sq / (float)P - mean * mean;
    stats[c]      = mean;
    stats[CH + c] = rsqrtf(var + 1e-5f);
}

// ---------------- BN apply + quantize -> padded NHWC bf16 slot ----------------
__global__ void bn_apply_k(const float* __restrict__ preBN, const float* __restrict__ stats,
                           const float* __restrict__ gamma, const float* __restrict__ beta,
                           unsigned short* __restrict__ dst, int H, int W) {
    const int Hp = H + 2, Wp = W + 2;
    long long idx = (long long)blockIdx.x * blockDim.x + threadIdx.x;
    long long total = 32LL * Hp * Wp * CH;
    if (idx >= total) return;
    int c = (int)(idx & (CH - 1));
    long long q = idx >> 8;
    int wp = (int)(q % Wp); q /= Wp;
    int hp = (int)(q % Hp);
    int n  = (int)(q / Hp);
    unsigned short out = 0;
    if (hp >= 1 && hp <= H && wp >= 1 && wp <= W) {
        long long p = ((long long)n * H + (hp - 1)) * W + (wp - 1);
        float v = preBN[(p << 8) + c];
        float y = gamma[c] * ((v - stats[c]) * stats[CH + c]) + beta[c];
        out = f2bf(quant64(y));
    }
    dst[idx] = out;
}

// ---------------- 2x2 maxpool on padded bf16 slots ----------------
__global__ void maxpool_k(const unsigned short* __restrict__ src, unsigned short* __restrict__ dst,
                          int H, int W) {
    const int Ho = H >> 1, Wo = W >> 1;
    const int Hop = Ho + 2, Wop = Wo + 2, Hp = H + 2, Wp = W + 2;
    long long idx = (long long)blockIdx.x * blockDim.x + threadIdx.x;
    long long total = 32LL * Hop * Wop * CH;
    if (idx >= total) return;
    int c = (int)(idx & (CH - 1));
    long long q = idx >> 8;
    int wo = (int)(q % Wop); q /= Wop;
    int ho = (int)(q % Hop);
    int n  = (int)(q / Hop);
    unsigned short out = 0;
    if (ho >= 1 && ho <= Ho && wo >= 1 && wo <= Wo) {
        int hs = (ho - 1) * 2 + 1, wsrc = (wo - 1) * 2 + 1;
        const unsigned short* b = src + ((((long long)n * Hp + hs) * Wp + wsrc) << 8) + c;
        long long rs = ((long long)Wp) << 8;
        float m = fmaxf(fmaxf(bf2f(b[0]), bf2f(b[CH])),
                        fmaxf(bf2f(b[rs]), bf2f(b[rs + CH])));
        out = f2bf(m);   // exact: max of bf16 values round-trips
    }
    dst[idx] = out;
}

// ---------------- global spatial max -> feat bf16 [32][256] ----------------
__global__ void feat_max_k(const unsigned short* __restrict__ src, unsigned short* __restrict__ feat,
                           int H, int W) {
    int id = blockIdx.x * blockDim.x + threadIdx.x;
    if (id >= 32 * CH) return;
    int c = id & (CH - 1);
    int n = id >> 8;
    const int Hp = H + 2, Wp = W + 2;
    float m = -3.4e38f;
    for (int h = 0; h < H; ++h)
        for (int w = 0; w < W; ++w)
            m = fmaxf(m, bf2f(src[((((long long)n * Hp + h + 1) * Wp) + w + 1) * CH + c]));
    feat[id] = f2bf(m);
}

// ---------------- classifier: feat[32x256] @ lin_w.T + b, WMMA ----------------
__global__ __launch_bounds__(256) void linear_wmma_k(
    const unsigned short* __restrict__ feat,   // [32][256] bf16
    const unsigned short* __restrict__ lw,     // [1000][256] bf16
    const float* __restrict__ lb,              // [1000] f32
    float* __restrict__ out)                   // [32][1000] f32
{
    const int lane = threadIdx.x & 31;
    const int wid  = blockIdx.x * (blockDim.x >> 5) + (threadIdx.x >> 5);
    if (wid >= 2 * 63) return;                 // 2 M-tiles x 63 N-tiles
    const int mt = wid / 63, nt = wid % 63;
    const int l15  = lane & 15;
    const int kOff = (lane >> 4) << 3;
    const int m  = mt * 16 + l15;
    const int n  = nt * 16 + l15;
    const int nc = (n < 1000) ? n : 999;       // clamp loads; store guarded

    v8f acc = {};
    #pragma unroll
    for (int kc = 0; kc < CH; kc += 32) {
        v16bf a = mkfrag(feat + m * CH + kc + kOff);
        v16bf b = mkfrag(lw + (long long)nc * CH + kc + kOff);
        acc = __builtin_amdgcn_wmma_f32_16x16x32_bf16(false, a, false, b, (short)0, acc, false, false);
    }
    const int rowM = kOff;
    #pragma unroll
    for (int j = 0; j < 8; ++j) {
        int mm = mt * 16 + rowM + j;
        int nn = nt * 16 + l15;
        if (nn < 1000) out[(long long)mm * 1000 + nn] = acc[j] + lb[nn];
    }
}

// ---------------- host orchestration (static, deterministic schedule) ----------------
extern "C" void kernel_launch(void* const* d_in, const int* in_sizes, int n_in,
                              void* d_out, int out_size, void* d_ws, size_t ws_size,
                              hipStream_t stream) {
    const float* x      = (const float*)d_in[0];   // (32,3,64,64)
    const float* conv_w = (const float*)d_in[1];   // (256,256,3,3)
    const float* lin_w  = (const float*)d_in[2];   // (1000,256)
    const float* lin_b  = (const float*)d_in[3];   // (1000,)
    const float* alpha  = (const float*)d_in[4];   // (10,6)
    const float* gamma  = (const float*)d_in[5];   // (10,256)
    const float* beta   = (const float*)d_in[6];   // (10,256)
    float* out = (float*)d_out;                    // (32,1000)

    // ---- workspace carve (static offsets) ----
    char* ws = (char*)d_ws;
    size_t off = 0;
    auto carve = [&](size_t bytes) -> void* {
        void* p = ws + off;
        off += (bytes + 255) & ~(size_t)255;
        return p;
    };
    const size_t SLOT_ELEMS = 32ULL * 66 * 66 * CH;   // padded NHWC bf16 @ max res
    unsigned short* slot[5];
    for (int i = 0; i < 5; ++i) slot[i] = (unsigned short*)carve(SLOT_ELEMS * 2);
    float*          preBN   = (float*)carve(32ULL * 64 * 64 * CH * 4);
    unsigned short* convW   = (unsigned short*)carve((size_t)CH * 9 * CH * 2);
    unsigned short* linW    = (unsigned short*)carve(1000ULL * CH * 2);
    unsigned short* feat    = (unsigned short*)carve(32ULL * CH * 2);
    float*          stats   = (float*)carve(2ULL * CH * 4);
    float*          partial = (float*)carve(64ULL * 512 * 4);

    // ---- one-time weight swizzles ----
    {
        int tw = CH * CH * 9;
        convw_to_bf_k<<<(tw + 255) / 256, 256, 0, stream>>>(conv_w, convW);
        int tl = 1000 * CH;
        f32_to_bf_k<<<(tl + 255) / 256, 256, 0, stream>>>(lin_w, linW, tl);
    }

    // ---- buffer free-list + history slots (hist[0] stays None, per reference) ----
    int freeStk[5] = {4, 3, 2, 1, 0};
    int top = 5;
    auto balloc = [&]() { return freeStk[--top]; };
    auto bfree  = [&](int i) { freeStk[top++] = i; };
    int hist[5] = {-1, -1, -1, -1, -1};

    int x0 = balloc();
    {
        long long tot = 32LL * 66 * 66 * CH;
        pad_quant_in_k<<<(int)((tot + 255) / 256), 256, 0, stream>>>(x, slot[x0]);
    }
    hist[4] = x0;

    int H = 64, lgW = 6;
    for (int t = 0; t < 10; ++t) {
        const int W = H;
        const int P = 32 * H * W;
        const unsigned short* hp[4];
        for (int k = 0; k < 4; ++k)
            hp[k] = (hist[k + 1] >= 0) ? slot[hist[k + 1]] : (const unsigned short*)nullptr;

        // conv + fused residual epilogue -> preBN  (2 pixel tiles / block)
        conv_wmma_k<<<P / 32, 256, 0, stream>>>(
            slot[hist[4]], convW, hp[0], hp[1], hp[2], hp[3],
            alpha + t * 6, preBN, lgW);

        // batchnorm stats (coalesced, deterministic) + apply + quantize -> new slot
        bn_partial_k<<<64, 256, 0, stream>>>(preBN, partial, P);
        bn_finalize_k<<<1, 256, 0, stream>>>(partial, stats, P);
        int nw = balloc();
        {
            long long tot = 32LL * (H + 2) * (W + 2) * CH;
            bn_apply_k<<<(int)((tot + 255) / 256), 256, 0, stream>>>(
                preBN, stats, gamma + t * CH, beta + t * CH, slot[nw], H, W);
        }

        // history shift (hist[0] untouched / always None)
        int dead = hist[1];
        hist[1] = hist[2]; hist[2] = hist[3]; hist[3] = hist[4]; hist[4] = nw;
        if (dead >= 0) bfree(dead);

        // pool every live history slot when t % 3 == 2
        if (t % 3 == 2) {
            for (int i = 1; i < 5; ++i) {
                if (hist[i] >= 0) {
                    int d = balloc();
                    long long tot = 32LL * (H / 2 + 2) * (W / 2 + 2) * CH;
                    maxpool_k<<<(int)((tot + 255) / 256), 256, 0, stream>>>(
                        slot[hist[i]], slot[d], H, W);
                    bfree(hist[i]);
                    hist[i] = d;
                }
            }
            H >>= 1; lgW -= 1;
        }
    }

    // global spatial max -> feat ; classifier GEMM -> out
    feat_max_k<<<32, 256, 0, stream>>>(slot[hist[4]], feat, H, H);
    linear_wmma_k<<<16, 256, 0, stream>>>(feat, linW, lin_b, out);
}
